// RSCM_31052613550401
// MI455X (gfx1250) — compile-verified, matrix operand order checked by reading, not run on hardware
//
#include <hip/hip_runtime.h>
#include <hip/hip_bf16.h>
#include <math.h>

// ---------------- problem constants ----------------
#define BATCH   4
#define CDIM    192
#define HEADS   6
#define CH      32
#define NPIX    65536      // 256*256
#define NCHUNK  64         // chunks of N per (b,head) in k_gram
#define CHUNKPX 1024
#define SUBT    16         // 16 subtiles of 64 px per chunk
#define TPX     64         // pixels per subtile

#define XST 200            // LDS stride for transposed x tile (bf16), mult of 8
#define QST 72             // LDS stride for staged Q/K tiles (bf16), mult of 8

typedef __attribute__((ext_vector_type(16))) __bf16 v16bf;
typedef __attribute__((ext_vector_type(8)))  float  v8f;

union BFrag { v16bf v; __bf16 e[16]; uint4 q[2]; };
union FAcc  { v8f  v; float  e[8];  };
union Pk    { __bf16 h[2]; unsigned int u; };

// ---------------- kernel 0: zero accumulators ----------------
__global__ void k_zero(float* __restrict__ p, int n) {
    int i = blockIdx.x * blockDim.x + threadIdx.x;
    if (i < n) p[i] = 0.f;
}

// ---------------- kernel 1: Gram matrices + row norms ----------------
// grid = BATCH*HEADS*NCHUNK blocks of 256 threads (8 waves)
__global__ __launch_bounds__(256) void k_gram(
        const float* __restrict__ x, const float* __restrict__ wqkv,
        float* __restrict__ G, float* __restrict__ NQ, float* __restrict__ NK)
{
    __shared__ __bf16 sXT[TPX * XST];    // x tile, transposed: [pixel][channel]
    __shared__ __bf16 sQ [CH  * QST];    // Q tile: [qchan][pixel]
    __shared__ __bf16 sK [CH  * QST];    // K tile
    __shared__ float  sRed[4 * 256];     // k-half reduction for Gram
    __shared__ float  sNQ[CH], sNK[CH];  // norm reduction

    const int tid  = threadIdx.x;
    const int wave = tid >> 5, lane = tid & 31;
    const int l16  = lane & 15, lh = lane >> 4;
    const int kb   = lh * 8;                       // WMMA per-lane K base

    const int bid   = blockIdx.x;
    const int chunk = bid & (NCHUNK - 1);
    const int head  = (bid >> 6) % HEADS;
    const int b     = bid / (NCHUNK * HEADS);

    const float* xb = x + (size_t)b * CDIM * NPIX;

    if (tid < CH) { sNQ[tid] = 0.f; sNK[tid] = 0.f; }

    // staging assignment: quads of (2 channels x 2 pixels); 12 quads/thread.
    // in-wave map: 8 pixel-pairs x 4 channel-pairs -> conflict-free b32 stores,
    // 64B-contiguous global reads.
    int qc[12], qn[12];                  // channel (even), pixel (even) per quad
    #pragma unroll
    for (int i = 0; i < 12; ++i) {
        const int g = i * 8 + wave;      // 0..95
        qn[i] = ((g & 3) * 8 + (lane & 7)) * 2;
        qc[i] = ((g >> 2) * 4 + (lane >> 3)) * 2;
    }
    float2 ra[12], rb[12];               // staged global data (double use)

    // ---- load Wq/Wk A-fragments once (kept in VGPRs for whole chunk) ----
    const int mi = wave & 1, ni = wave >> 1;       // Q/K output tile of this wave
    BFrag aQ[6], aK[6];
    {
        const int row = head * CH + mi * 16 + l16;             // q rows 0..191
        const float* wq = wqkv + (size_t)row * CDIM;
        const float* wk = wqkv + (size_t)(CDIM + row) * CDIM;  // k rows 192..383
        #pragma unroll
        for (int kt = 0; kt < 6; ++kt) {
            #pragma unroll
            for (int j = 0; j < 16; ++j) {
                const int k = kt * 32 + kb + j + ((j >= 8) ? 8 : 0);
                aQ[kt].e[j] = (__bf16)wq[k];
                aK[kt].e[j] = (__bf16)wk[k];
            }
        }
    }

    // Gram tile ownership: (gm,gn) tile, kh = n-half of the contraction
    const int gm = wave & 1, gn = (wave >> 1) & 1, kh = wave >> 2;
    FAcc gacc;
    #pragma unroll
    for (int v = 0; v < 8; ++v) gacc.e[v] = 0.f;

    float vnq[8], vnk[8];                // per-lane norm partials (from regs)
    #pragma unroll
    for (int v = 0; v < 8; ++v) { vnq[v] = 0.f; vnk[v] = 0.f; }

    const int base0 = chunk * CHUNKPX;
    const int px = ni * 16 + l16;

    // prefetch subtile 0
    #pragma unroll
    for (int i = 0; i < 12; ++i) {
        ra[i] = *(const float2*)&xb[(size_t)qc[i] * NPIX + base0 + qn[i]];
        rb[i] = *(const float2*)&xb[(size_t)(qc[i] + 1) * NPIX + base0 + qn[i]];
    }

    for (int s = 0; s < SUBT; ++s) {
        __syncthreads();                 // prev iter's sXT readers done
        // pack f32->bf16 channel pairs, one b32 store per pixel
        #pragma unroll
        for (int i = 0; i < 12; ++i) {
            Pk p0, p1;
            p0.h[0] = (__bf16)ra[i].x; p0.h[1] = (__bf16)rb[i].x;
            p1.h[0] = (__bf16)ra[i].y; p1.h[1] = (__bf16)rb[i].y;
            *(unsigned int*)&sXT[qn[i] * XST + qc[i]]       = p0.u;
            *(unsigned int*)&sXT[(qn[i] + 1) * XST + qc[i]] = p1.u;
        }
        __syncthreads();
        // prefetch next subtile (overlaps with all WMMA work below)
        if (s + 1 < SUBT) {
            const int nb = base0 + (s + 1) * TPX;
            #pragma unroll
            for (int i = 0; i < 12; ++i) {
                ra[i] = *(const float2*)&xb[(size_t)qc[i] * NPIX + nb + qn[i]];
                rb[i] = *(const float2*)&xb[(size_t)(qc[i] + 1) * NPIX + nb + qn[i]];
            }
        }

        // ---- Q,K tiles: (32 x 64) = Wq/Wk(32x192) @ X(192x64) ----
        FAcc accq, acck;
        #pragma unroll
        for (int v = 0; v < 8; ++v) { accq.e[v] = 0.f; acck.e[v] = 0.f; }
        #pragma unroll
        for (int kt = 0; kt < 6; ++kt) {
            BFrag bx;
            const uint4* p = (const uint4*)&sXT[px * XST + kt * 32 + kb];
            bx.q[0] = p[0];          // K = kt*32+kb   .. +7
            bx.q[1] = p[2];          // K = kt*32+kb+16.. +23
            accq.v = __builtin_amdgcn_wmma_f32_16x16x32_bf16(
                         false, aQ[kt].v, false, bx.v, (short)0, accq.v, false, false);
            acck.v = __builtin_amdgcn_wmma_f32_16x16x32_bf16(
                         false, aK[kt].v, false, bx.v, (short)0, acck.v, false, false);
        }
        // norms straight from accumulator registers
        #pragma unroll
        for (int v = 0; v < 8; ++v) {
            vnq[v] += accq.e[v] * accq.e[v];
            vnk[v] += acck.e[v] * acck.e[v];
        }
        // restage Q,K as bf16 for the Gram WMMA
        #pragma unroll
        for (int v = 0; v < 8; ++v) {
            const int row = mi * 16 + v + 8 * lh;
            sQ[row * QST + px] = (__bf16)accq.e[v];
            sK[row * QST + px] = (__bf16)acck.e[v];
        }
        __syncthreads();

        // ---- Gram: G(32x32) += Q(32x64) @ K^T(64x32), one WMMA per wave ----
        {
            BFrag ga, gb;
            const uint4* pa = (const uint4*)&sQ[(gm * 16 + l16) * QST + kh * 32 + kb];
            const uint4* pb = (const uint4*)&sK[(gn * 16 + l16) * QST + kh * 32 + kb];
            ga.q[0] = pa[0]; ga.q[1] = pa[2];
            gb.q[0] = pb[0]; gb.q[1] = pb[2];
            gacc.v = __builtin_amdgcn_wmma_f32_16x16x32_bf16(
                         false, ga.v, false, gb.v, (short)0, gacc.v, false, false);
        }
    }

    // ---- reductions ----
    __syncthreads();
    const int tile = gm + gn * 2;
    if (kh == 1) {
        #pragma unroll
        for (int v = 0; v < 8; ++v) sRed[tile * 256 + v * 32 + lane] = gacc.e[v];
    }
    #pragma unroll
    for (int v = 0; v < 8; ++v) {
        const int row = mi * 16 + v + 8 * lh;
        atomicAdd(&sNQ[row], vnq[v]);
        atomicAdd(&sNK[row], vnk[v]);
    }
    __syncthreads();
    if (kh == 0) {
        float* Gb = G + (size_t)(b * HEADS + head) * CH * CH;
        #pragma unroll
        for (int v = 0; v < 8; ++v) {
            const float val = gacc.e[v] + sRed[tile * 256 + v * 32 + lane];
            const int row = gm * 16 + v + 8 * lh;
            const int col = gn * 16 + l16;
            atomicAdd(&Gb[row * CH + col], val);
        }
    }
    if (tid < CH) {
        atomicAdd(&NQ[b * CDIM + head * CH + tid], sNQ[tid]);
        atomicAdd(&NK[b * CDIM + head * CH + tid], sNK[tid]);
    }
}

// ---------------- kernel 2: softmax + fuse W_proj @ A @ W_v ----------------
// grid = BATCH blocks of 256 threads
__global__ __launch_bounds__(256) void k_finalize(
        const float* __restrict__ G, const float* __restrict__ NQ,
        const float* __restrict__ NK, const float* __restrict__ wqkv,
        const float* __restrict__ bqkv, const float* __restrict__ wproj,
        const float* __restrict__ bproj, const float* __restrict__ temp,
        __bf16* __restrict__ MF, float* __restrict__ BIASF)
{
    __shared__ float sAttn[CDIM * CH];   // 24 KB
    __shared__ float sAVB[CDIM];
    __shared__ float sAVc[CDIM * 32];    // 24 KB  (x-chunk of A@Wv)
    const int t = threadIdx.x;
    const int b = blockIdx.x;
    const float EPSV = 1e-12f;

    if (t < CDIM) {                       // one attention row per thread
        const int h = t >> 5;
        const float nq  = fmaxf(sqrtf(NQ[b * CDIM + t]), EPSV);
        const float tmp = temp[h];
        const float* grow = G + ((size_t)(b * HEADS + h) * CH + (t & 31)) * CH;
        float vals[CH]; float mx = -1e30f;
        for (int d = 0; d < CH; ++d) {
            const float nk = fmaxf(sqrtf(NK[b * CDIM + h * CH + d]), EPSV);
            const float v  = -tmp * grow[d] / (nq * nk);
            vals[d] = v; mx = fmaxf(mx, v);
        }
        float sum = 0.f;
        for (int d = 0; d < CH; ++d) { vals[d] = __expf(vals[d] - mx); sum += vals[d]; }
        const float inv = 1.f / sum;
        float avb = 0.f;
        for (int d = 0; d < CH; ++d) {
            const float a = vals[d] * inv;
            sAttn[t * CH + d] = a;
            avb += a * bqkv[2 * CDIM + h * CH + d];
        }
        sAVB[t] = avb;
    }
    __syncthreads();

    for (int xc = 0; xc < 6; ++xc) {      // 32-wide column chunks of 192
        for (int i = 0; i < (CDIM * 32) / 256; ++i) {   // AV chunk = A @ Wv
            const int idx = i * 256 + t;
            const int cg = idx >> 5, xx = idx & 31, h = cg >> 5;
            float s = 0.f;
            for (int d = 0; d < CH; ++d)
                s += sAttn[cg * CH + d] *
                     wqkv[(size_t)(2 * CDIM + h * CH + d) * CDIM + xc * 32 + xx];
            sAVc[cg * 32 + xx] = s;
        }
        __syncthreads();
        for (int i = 0; i < (CDIM * 32) / 256; ++i) {   // Mf = Wproj @ AV
            const int idx = i * 256 + t;
            const int o = idx >> 5, xx = idx & 31;
            float s = 0.f;
            for (int cg = 0; cg < CDIM; ++cg)
                s += wproj[(size_t)o * CDIM + cg] * sAVc[cg * 32 + xx];
            MF[((size_t)b * CDIM + o) * CDIM + xc * 32 + xx] = (__bf16)s;
        }
        __syncthreads();
    }
    if (t < CDIM) {
        float s = 0.f;
        for (int cg = 0; cg < CDIM; ++cg)
            s += wproj[(size_t)t * CDIM + cg] * sAVB[cg];
        BIASF[b * CDIM + t] = bproj[t] + s;
    }
}

// ---------------- kernel 3: y = Mfused @ x + bias ----------------
#define K3_SUBT  8
__global__ __launch_bounds__(384) void k_out(
        const float* __restrict__ x, const __bf16* __restrict__ MF,
        const float* __restrict__ BIASF, float* __restrict__ y)
{
    __shared__ __bf16 sXT[TPX * XST];
    const int tid = threadIdx.x, wave = tid >> 5, lane = tid & 31;
    const int l16 = lane & 15, lh = lane >> 4, kb = lh * 8;
    const int nblk = NPIX / (TPX * K3_SUBT);       // 128
    const int blk  = blockIdx.x % nblk;
    const int b    = blockIdx.x / nblk;

    const float*  xb  = x  + (size_t)b * CDIM * NPIX;
    float*        yb  = y  + (size_t)b * CDIM * NPIX;
    const __bf16* mfb = MF + (size_t)b * CDIM * CDIM;

    // staging quads: 8 per thread (12 waves)
    int qc[8], qn[8];
    #pragma unroll
    for (int i = 0; i < 8; ++i) {
        const int g = i * 12 + wave;     // 0..95
        qn[i] = ((g & 3) * 8 + (lane & 7)) * 2;
        qc[i] = ((g >> 2) * 4 + (lane >> 3)) * 2;
    }
    float2 ra[8], rb[8];

    // fused-matrix A-fragments for this wave's 16 output rows (loaded once)
    BFrag aM[6];
    {
        const __bf16* mrow = mfb + (size_t)(wave * 16 + l16) * CDIM;
        #pragma unroll
        for (int kt = 0; kt < 6; ++kt) {
            const uint4* p = (const uint4*)&mrow[kt * 32 + kb];
            aM[kt].q[0] = p[0]; aM[kt].q[1] = p[2];
        }
    }
    float bias[8];
    #pragma unroll
    for (int v = 0; v < 8; ++v)
        bias[v] = BIASF[b * CDIM + wave * 16 + v + 8 * lh];

    const int base0 = blk * (TPX * K3_SUBT);
    #pragma unroll
    for (int i = 0; i < 8; ++i) {
        ra[i] = *(const float2*)&xb[(size_t)qc[i] * NPIX + base0 + qn[i]];
        rb[i] = *(const float2*)&xb[(size_t)(qc[i] + 1) * NPIX + base0 + qn[i]];
    }

    for (int s = 0; s < K3_SUBT; ++s) {
        const int base = base0 + s * TPX;
        __syncthreads();
        #pragma unroll
        for (int i = 0; i < 8; ++i) {
            Pk p0, p1;
            p0.h[0] = (__bf16)ra[i].x; p0.h[1] = (__bf16)rb[i].x;
            p1.h[0] = (__bf16)ra[i].y; p1.h[1] = (__bf16)rb[i].y;
            *(unsigned int*)&sXT[qn[i] * XST + qc[i]]       = p0.u;
            *(unsigned int*)&sXT[(qn[i] + 1) * XST + qc[i]] = p1.u;
        }
        __syncthreads();
        if (s + 1 < K3_SUBT) {
            const int nb = base0 + (s + 1) * TPX;
            #pragma unroll
            for (int i = 0; i < 8; ++i) {
                ra[i] = *(const float2*)&xb[(size_t)qc[i] * NPIX + nb + qn[i]];
                rb[i] = *(const float2*)&xb[(size_t)(qc[i] + 1) * NPIX + nb + qn[i]];
            }
        }
        #pragma unroll
        for (int ni = 0; ni < 4; ++ni) {
            FAcc acc;
            #pragma unroll
            for (int v = 0; v < 8; ++v) acc.e[v] = 0.f;
            const int px = ni * 16 + l16;
            #pragma unroll
            for (int kt = 0; kt < 6; ++kt) {
                BFrag bx;
                const uint4* p = (const uint4*)&sXT[px * XST + kt * 32 + kb];
                bx.q[0] = p[0]; bx.q[1] = p[2];
                acc.v = __builtin_amdgcn_wmma_f32_16x16x32_bf16(
                            false, aM[kt].v, false, bx.v, (short)0, acc.v, false, false);
            }
            #pragma unroll
            for (int v = 0; v < 8; ++v) {
                const int row = wave * 16 + v + 8 * lh;
                yb[(size_t)row * NPIX + base + px] = acc.e[v] + bias[v];
            }
        }
    }
}

// ---------------- launch ----------------
extern "C" void kernel_launch(void* const* d_in, const int* in_sizes, int n_in,
                              void* d_out, int out_size, void* d_ws, size_t ws_size,
                              hipStream_t stream) {
    const float* x     = (const float*)d_in[0];
    const float* wqkv  = (const float*)d_in[1];
    const float* bqkv  = (const float*)d_in[2];
    const float* wproj = (const float*)d_in[3];
    const float* bproj = (const float*)d_in[4];
    const float* temp  = (const float*)d_in[5];
    float* out = (float*)d_out;

    // workspace layout (16B aligned throughout)
    float*  G     = (float*)d_ws;                         // 24576
    float*  NQ    = G + BATCH * HEADS * CH * CH;          // 768
    float*  NK    = NQ + BATCH * CDIM;                    // 768
    float*  BIASF = NK + BATCH * CDIM;                    // 768
    __bf16* MF    = (__bf16*)(BIASF + BATCH * CDIM);      // 4*192*192 bf16

    const int nzero = BATCH * HEADS * CH * CH + 2 * BATCH * CDIM;  // 26112
    k_zero<<<(nzero + 255) / 256, 256, 0, stream>>>(G, nzero);
    k_gram<<<BATCH * HEADS * NCHUNK, 256, 0, stream>>>(x, wqkv, G, NQ, NK);
    k_finalize<<<BATCH, 256, 0, stream>>>(G, NQ, NK, wqkv, bqkv, wproj, bproj,
                                          temp, MF, BIASF);
    k_out<<<BATCH * (NPIX / (TPX * K3_SUBT)), 384, 0, stream>>>(x, MF, BIASF, out);
}
